// SceneEncoder_2499670966423
// MI455X (gfx1250) — compile-verified, matrix operand order checked by reading, not run on hardware
//
// SceneEncoder for MI455X (gfx1250): bf16 WMMA for all large GEMMs, fused
// Performer kv-reduction (never materializes K/V: saves 536MB of traffic),
// deterministic fixed-order reductions, f32 accumulate everywhere.
#include <hip/hip_runtime.h>
#include <stdint.h>

typedef unsigned short u16; // bf16 storage
typedef __attribute__((ext_vector_type(16))) __bf16 v16bf;
typedef __attribute__((ext_vector_type(8)))  float  v8f;

#define LN_EPS 0.001f
#define STAB   0.001f
#define OBJ_NORM 0.35355339059327373f  // 64^-0.25
#define RG_NORM  0.70710678118654752f  // 4^-0.25

union FragAB { v16bf v; uint4 q[2]; };
union FragC  { v8f  v; float f[8]; };

__device__ __forceinline__ u16 f2bf(float x) {
  union { float f; unsigned u; } a; a.f = x;
  unsigned r = a.u + 0x7FFFu + ((a.u >> 16) & 1u);   // RNE
  return (u16)(r >> 16);
}
__device__ __forceinline__ float bf2f(u16 h) {
  union { unsigned u; float f; } a; a.u = ((unsigned)h) << 16; return a.f;
}

// A-fragment (16x32 bf16): lane L holds row (base+ (L&15)), K chunks
// [k0+off, +8) and [k0+off+16, +8) with off = (L>>4)*8.  B-fragments are the
// same pattern applied to the pre-transposed (NxK) weight matrix.
__device__ __forceinline__ v16bf load_frag(const u16* __restrict__ base,
                                           long long ld, long long row0, int k0) {
  const int lane = threadIdx.x & 31;
  const u16* p = base + (row0 + (lane & 15)) * ld + k0 + ((lane >> 4) << 3);
  FragAB f;
  f.q[0] = *reinterpret_cast<const uint4*>(p);
  f.q[1] = *reinterpret_cast<const uint4*>(p + 16);
  return f.v;
}

// ---------------------------------------------------------------------------
// Generic GEMM: C[M,N] = act(A[MxK] @ Bt[NxK]^T + bias)
// act: 0=none, 1=relu, 2=performer (relu(v*scale)+STAB)
// Block 256 thr = 8 waves (2Mx4N), wave tile 32x32 -> block tile 64x128.
// ---------------------------------------------------------------------------
__global__ __launch_bounds__(256)
void wmma_gemm_kernel(const u16* __restrict__ A, const u16* __restrict__ Bt,
                      const float* __restrict__ bias,
                      float* __restrict__ Cf, u16* __restrict__ Cbf,
                      int M, int N, int K, int act, float scale)
{
  const int wave = threadIdx.x >> 5;
  const int lane = threadIdx.x & 31;
  const int m0 = blockIdx.y * 64 + (wave & 1) * 32;
  const int n0 = blockIdx.x * 128 + (wave >> 1) * 32;

  v8f c00 = {}, c01 = {}, c10 = {}, c11 = {};
  for (int k = 0; k < K; k += 32) {
    if (k + 32 < K) { // hide HBM latency on the streamed A operand
      __builtin_prefetch(A + (long long)(m0 + (lane & 15)) * K + k + 32, 0, 1);
      __builtin_prefetch(Bt + (long long)(n0 + (lane & 15)) * K + k + 32, 0, 1);
    }
    v16bf a0 = load_frag(A,  K, m0,      k);
    v16bf a1 = load_frag(A,  K, m0 + 16, k);
    v16bf b0 = load_frag(Bt, K, n0,      k);
    v16bf b1 = load_frag(Bt, K, n0 + 16, k);
    c00 = __builtin_amdgcn_wmma_f32_16x16x32_bf16(false, a0, false, b0, (short)0, c00, false, false);
    c01 = __builtin_amdgcn_wmma_f32_16x16x32_bf16(false, a0, false, b1, (short)0, c01, false, false);
    c10 = __builtin_amdgcn_wmma_f32_16x16x32_bf16(false, a1, false, b0, (short)0, c10, false, false);
    c11 = __builtin_amdgcn_wmma_f32_16x16x32_bf16(false, a1, false, b1, (short)0, c11, false, false);
  }
  v8f cs[2][2] = {{c00, c01}, {c10, c11}};
  for (int im = 0; im < 2; ++im)
    for (int in = 0; in < 2; ++in) {
      const int ncol  = n0 + in * 16 + (lane & 15);
      const int rbase = m0 + im * 16 + ((lane >> 4) << 3);
      const float bv = bias ? bias[ncol] : 0.f;
      FragC fc; fc.v = cs[im][in];
#pragma unroll
      for (int r = 0; r < 8; ++r) {
        float v = fc.f[r] + bv;
        if (act == 1)      v = fmaxf(v, 0.f);
        else if (act == 2) v = fmaxf(v * scale, 0.f) + STAB;
        const long long idx = (long long)(rbase + r) * N + ncol;
        if (Cf)  Cf[idx]  = v;
        if (Cbf) Cbf[idx] = f2bf(v);
      }
    }
}

// ---------------------------------------------------------------------------
// Fused Performer K/V projection + kv reduction for the object stage.
// grid (h=16, chunk=16, b=8); each block covers 512 rows of rg_out for one
// head: waves 0-3 project kp (relu(k*norm)+STAB), waves 4-7 project v; tiles
// land in LDS and are reduced into kv[64x64] / ksum[64] register partials.
// ---------------------------------------------------------------------------
__global__ __launch_bounds__(256)
void kv_partial_kernel(const u16* __restrict__ rg_out,  // [65536,1024]
                       const u16* __restrict__ Wkt,     // [1024,1024] (NxK)
                       const u16* __restrict__ Wvt,
                       float* __restrict__ kv_part,     // [8][16][16][64][64]
                       float* __restrict__ ksum_part,   // [8][16][16][64]
                       float scale)
{
  const int h = blockIdx.x, chunk = blockIdx.y, b = blockIdx.z;
  const int wave = threadIdx.x >> 5;
  const int lane = threadIdx.x & 31;
  const int wm   = wave & 3;       // 32-row strip within 128-row subtile
  const int wsel = wave >> 2;      // 0 -> kp, 1 -> v
  const u16* Bt = (wsel == 0 ? Wkt : Wvt) + (long long)h * 64 * 1024;

  __shared__ float lds_kp[128][64];
  __shared__ float lds_v [128][64];

  const int tid = threadIdx.x;
  const int m_elem = tid & 63;     // kv row index m
  const int dgrp   = tid >> 6;     // d-range = dgrp*16 .. +15
  float kv_acc[16];
#pragma unroll
  for (int e = 0; e < 16; ++e) kv_acc[e] = 0.f;
  float ksum_acc = 0.f;

  const long long rowBase = (long long)b * 8192 + (long long)chunk * 512;

  for (int sub = 0; sub < 4; ++sub) {
    const long long aRow = rowBase + sub * 128 + wm * 32;
    v8f c[2][4] = {};
    for (int k = 0; k < 1024; k += 32) {
      if (k + 32 < 1024)
        __builtin_prefetch(rg_out + (aRow + (lane & 15)) * 1024 + k + 32, 0, 1);
      v16bf a0 = load_frag(rg_out, 1024, aRow,      k);
      v16bf a1 = load_frag(rg_out, 1024, aRow + 16, k);
      v16bf b0 = load_frag(Bt, 1024,  0, k);
      v16bf b1 = load_frag(Bt, 1024, 16, k);
      v16bf b2 = load_frag(Bt, 1024, 32, k);
      v16bf b3 = load_frag(Bt, 1024, 48, k);
      c[0][0] = __builtin_amdgcn_wmma_f32_16x16x32_bf16(false, a0, false, b0, (short)0, c[0][0], false, false);
      c[0][1] = __builtin_amdgcn_wmma_f32_16x16x32_bf16(false, a0, false, b1, (short)0, c[0][1], false, false);
      c[0][2] = __builtin_amdgcn_wmma_f32_16x16x32_bf16(false, a0, false, b2, (short)0, c[0][2], false, false);
      c[0][3] = __builtin_amdgcn_wmma_f32_16x16x32_bf16(false, a0, false, b3, (short)0, c[0][3], false, false);
      c[1][0] = __builtin_amdgcn_wmma_f32_16x16x32_bf16(false, a1, false, b0, (short)0, c[1][0], false, false);
      c[1][1] = __builtin_amdgcn_wmma_f32_16x16x32_bf16(false, a1, false, b1, (short)0, c[1][1], false, false);
      c[1][2] = __builtin_amdgcn_wmma_f32_16x16x32_bf16(false, a1, false, b2, (short)0, c[1][2], false, false);
      c[1][3] = __builtin_amdgcn_wmma_f32_16x16x32_bf16(false, a1, false, b3, (short)0, c[1][3], false, false);
    }
    float* dst = (wsel == 0) ? &lds_kp[0][0] : &lds_v[0][0];
    for (int im = 0; im < 2; ++im)
      for (int jn = 0; jn < 4; ++jn) {
        const int ncol  = jn * 16 + (lane & 15);
        const int rbase = wm * 32 + im * 16 + ((lane >> 4) << 3);
        FragC fc; fc.v = c[im][jn];
#pragma unroll
        for (int r = 0; r < 8; ++r) {
          float v = fc.f[r];
          if (wsel == 0) v = fmaxf(v * scale, 0.f) + STAB;
          dst[(rbase + r) * 64 + ncol] = v;
        }
      }
    __syncthreads();
    for (int kk = 0; kk < 128; ++kk) {
      const float kpv = lds_kp[kk][m_elem];
      if (dgrp == 0) ksum_acc += kpv;
      const float* vrow = &lds_v[kk][dgrp * 16];
#pragma unroll
      for (int e = 0; e < 16; ++e) kv_acc[e] += kpv * vrow[e];
    }
    __syncthreads();
  }
  const long long base =
      ((((long long)b * 16 + chunk) * 16 + h) * 64 + m_elem) * 64 + dgrp * 16;
#pragma unroll
  for (int e = 0; e < 16; ++e) kv_part[base + e] = kv_acc[e];
  if (dgrp == 0)
    ksum_part[(((long long)b * 16 + chunk) * 16 + h) * 64 + m_elem] = ksum_acc;
}

__global__ __launch_bounds__(256)
void kv_reduce_kernel(const float* __restrict__ kv_part,
                      const float* __restrict__ ksum_part,
                      float* __restrict__ kv, float* __restrict__ ksum)
{
  const int bh = blockIdx.x, b = bh >> 4, h = bh & 15;
  for (int e = threadIdx.x; e < 4096; e += 256) {
    float s = 0.f;
    for (int c = 0; c < 16; ++c)
      s += kv_part[(((long long)b * 16 + c) * 16 + h) * 4096 + e];
    kv[(long long)bh * 4096 + e] = s;
  }
  for (int e = threadIdx.x; e < 64; e += 256) {
    float s = 0.f;
    for (int c = 0; c < 16; ++c)
      s += ksum_part[(((long long)b * 16 + c) * 16 + h) * 64 + e];
    ksum[bh * 64 + e] = s;
  }
}

// out[row, h*64+d] = (sum_m qp*kv) / (sum_m qp*ksum), emitted bf16 for Wo GEMM
__global__ __launch_bounds__(256)
void attn_combine_kernel(const float* __restrict__ qp,   // [1024,1024]
                         const float* __restrict__ kv,   // [8,16,64,64]
                         const float* __restrict__ ksum, // [8,16,64]
                         u16* __restrict__ outbf)        // [1024,1024]
{
  const int row = blockIdx.x, b = row >> 7;
  __shared__ float q_s[1024];
  for (int j = threadIdx.x; j < 1024; j += 256)
    q_s[j] = qp[(long long)row * 1024 + j];
  __syncthreads();
  for (int col = threadIdx.x; col < 1024; col += 256) {
    const int h = col >> 6, d = col & 63;
    const float* qh  = &q_s[h * 64];
    const float* kvh = &kv[(long long)(b * 16 + h) * 4096 + d];
    const float* ks  = &ksum[(b * 16 + h) * 64];
    float num = 0.f, den = 0.f;
    for (int m = 0; m < 64; ++m) { float q = qh[m]; num += q * kvh[m * 64]; den += q * ks[m]; }
    outbf[(long long)row * 1024 + col] = f2bf(num / den);
  }
}

// out = LN(a + b) with gain/beta; writes f32 and optional bf16 copy.
__global__ __launch_bounds__(256)
void ln_res_kernel(const float* __restrict__ a, const float* __restrict__ bres,
                   const float* __restrict__ g, const float* __restrict__ beta,
                   float* __restrict__ of, u16* __restrict__ obf)
{
  const int row = blockIdx.x, D = 1024;
  __shared__ float buf[1024];
  __shared__ float part[256];
  float s = 0.f;
  for (int j = threadIdx.x; j < D; j += 256) {
    float v = a[(long long)row * D + j] + bres[(long long)row * D + j];
    buf[j] = v; s += v;
  }
  part[threadIdx.x] = s; __syncthreads();
  for (int o = 128; o > 0; o >>= 1) {
    if (threadIdx.x < o) part[threadIdx.x] += part[threadIdx.x + o];
    __syncthreads();
  }
  const float mean = part[0] / D; __syncthreads();
  float vs = 0.f;
  for (int j = threadIdx.x; j < D; j += 256) { float d = buf[j] - mean; vs += d * d; }
  part[threadIdx.x] = vs; __syncthreads();
  for (int o = 128; o > 0; o >>= 1) {
    if (threadIdx.x < o) part[threadIdx.x] += part[threadIdx.x + o];
    __syncthreads();
  }
  const float inv = rsqrtf(part[0] / D + LN_EPS);
  for (int j = threadIdx.x; j < D; j += 256) {
    float o = (buf[j] - mean) * inv * g[j] + beta[j];
    of[(long long)row * D + j] = o;
    if (obf) obf[(long long)row * D + j] = f2bf(o);
  }
}

// W[KxN] f32 -> Wt[NxKpad] bf16 (zero-padded K)
__global__ void transpose_cvt_kernel(const float* __restrict__ W, u16* __restrict__ Wt,
                                     int K, int N, int Kpad)
{
  long long idx = (long long)blockIdx.x * 256 + threadIdx.x;
  if (idx >= (long long)N * Kpad) return;
  int n = (int)(idx / Kpad), k = (int)(idx % Kpad);
  float v = (k < K) ? W[(long long)k * N + n] : 0.f;
  Wt[idx] = f2bf(v);
}

// X[rows,D] f32 -> Y[rows,Dpad] bf16 (zero-padded)
__global__ void cvt_pad_kernel(const float* __restrict__ X, u16* __restrict__ Y,
                               int D, int Dpad, long long rows)
{
  long long idx = (long long)blockIdx.x * 256 + threadIdx.x;
  if (idx >= rows * Dpad) return;
  int d = (int)(idx % Dpad); long long r = idx / Dpad;
  float v = (d < D) ? X[r * D + d] : 0.f;
  Y[idx] = f2bf(v);
}

__global__ void add_pos_kernel(const float* __restrict__ xin, const float* __restrict__ pos,
                               float* __restrict__ xf, u16* __restrict__ xbf)
{
  long long idx = (long long)blockIdx.x * 256 + threadIdx.x;
  if (idx >= 1024LL * 1024) return;
  int col = (int)(idx & 1023); int t = (int)((idx >> 10) & 127);
  float v = xin[idx] + pos[t * 1024 + col];
  xf[idx] = v; xbf[idx] = f2bf(v);
}

// ------------------------- road-graph stage (16-wide, VALU) ----------------
__global__ void rg_dense_kernel(const float* __restrict__ road, const float* __restrict__ W,
                                const float* __restrict__ b, float* __restrict__ x)
{
  long long idx = (long long)blockIdx.x * 256 + threadIdx.x;
  if (idx >= 65536LL * 16) return;
  int j = (int)(idx & 15); long long row = idx >> 4;
  float s = b[j];
  const float* r = road + row * 8;
#pragma unroll
  for (int f = 0; f < 8; ++f) s += r[f] * W[f * 16 + j];
  x[idx] = fmaxf(s, 0.f);
}

__global__ __launch_bounds__(256)
void rg_attnA_kernel(const float* __restrict__ x,
                     const float* __restrict__ qkvW, const float* __restrict__ qkvB,
                     const float* __restrict__ Wq, const float* __restrict__ Wk,
                     const float* __restrict__ Wv,
                     float* __restrict__ qp_out, float* __restrict__ v_out,
                     float* __restrict__ part)  // [256 blocks][80]
{
  const long long row = (long long)blockIdx.x * 256 + threadIdx.x;
  float xr[16], q16[16], qp[16], kp[16], vv[16];
#pragma unroll
  for (int j = 0; j < 16; ++j) xr[j] = x[row * 16 + j];
#pragma unroll
  for (int j = 0; j < 16; ++j) {
    float s = qkvB[j];
    for (int c = 0; c < 16; ++c) s += xr[c] * qkvW[c * 48 + j];
    q16[j] = fmaxf(s, 0.f);
  }
#pragma unroll
  for (int j = 0; j < 16; ++j) {
    float sq = 0.f, sk = 0.f, sv = 0.f;
    for (int c = 0; c < 16; ++c) {
      float xv = q16[c];
      sq += xv * Wq[c * 16 + j]; sk += xv * Wk[c * 16 + j]; sv += xv * Wv[c * 16 + j];
    }
    qp[j] = fmaxf(sq * RG_NORM, 0.f) + STAB;
    kp[j] = fmaxf(sk * RG_NORM, 0.f) + STAB;
    vv[j] = sv;
  }
#pragma unroll
  for (int j = 0; j < 16; ++j) { qp_out[row * 16 + j] = qp[j]; v_out[row * 16 + j] = vv[j]; }

  __shared__ float skp[256][16];
  __shared__ float sv_[256][16];
#pragma unroll
  for (int j = 0; j < 16; ++j) { skp[threadIdx.x][j] = kp[j]; sv_[threadIdx.x][j] = vv[j]; }
  __syncthreads();
  // deterministic block reduction: elements 0..63 = kv[h][m][d], 64..79 = ksum
  if (threadIdx.x < 80) {
    const int e = threadIdx.x;
    float s = 0.f;
    if (e < 64) {
      const int mi = (e >> 4) * 4 + ((e >> 2) & 3);
      const int di = (e >> 4) * 4 + (e & 3);
      for (int r = 0; r < 256; ++r) s += skp[r][mi] * sv_[r][di];
    } else {
      const int j = e - 64;
      for (int r = 0; r < 256; ++r) s += skp[r][j];
    }
    part[(long long)blockIdx.x * 80 + e] = s;
  }
}

__global__ void rg_reduce_kernel(const float* __restrict__ part, float* __restrict__ kvk)
{
  const int b = blockIdx.x, e = threadIdx.x;
  if (e < 80) {
    float s = 0.f;
    for (int c = 0; c < 32; ++c) s += part[(long long)(b * 32 + c) * 80 + e];
    kvk[b * 80 + e] = s;
  }
}

__global__ void rg_attnC_kernel(const float* __restrict__ xin, const float* __restrict__ qp,
                                const float* __restrict__ kvk, const float* __restrict__ Wo,
                                const float* __restrict__ lng, const float* __restrict__ lnb,
                                const float* __restrict__ fcW, const float* __restrict__ fcb,
                                float* __restrict__ xout)
{
  long long row = (long long)blockIdx.x * 256 + threadIdx.x;
  if (row >= 65536) return;
  const int b = (int)(row >> 13);
  const float* kv = kvk + b * 80;
  const float* ks = kv + 64;
  float qpr[16], outv[16], attn[16], t[16], n_[16], y[16], o[16];
#pragma unroll
  for (int j = 0; j < 16; ++j) qpr[j] = qp[row * 16 + j];
  for (int h = 0; h < 4; ++h) {
    float den = 0.f;
    for (int m = 0; m < 4; ++m) den += qpr[h * 4 + m] * ks[h * 4 + m];
    for (int d = 0; d < 4; ++d) {
      float num = 0.f;
      for (int m = 0; m < 4; ++m) num += qpr[h * 4 + m] * kv[(h * 4 + m) * 4 + d];
      outv[h * 4 + d] = num / den;
    }
  }
#pragma unroll
  for (int e = 0; e < 16; ++e) {
    float s = 0.f;
    for (int j = 0; j < 16; ++j) s += outv[j] * Wo[j * 16 + e];
    attn[e] = s;
  }
#pragma unroll
  for (int e = 0; e < 16; ++e) t[e] = attn[e] + xin[row * 16 + e];

  float mu = 0.f; for (int e = 0; e < 16; ++e) mu += t[e]; mu *= (1.f / 16.f);
  float var = 0.f; for (int e = 0; e < 16; ++e) { float d = t[e] - mu; var += d * d; }
  float inv = rsqrtf(var * (1.f / 16.f) + LN_EPS);
  for (int e = 0; e < 16; ++e) n_[e] = (t[e] - mu) * inv * lng[e] + lnb[e];

#pragma unroll
  for (int e = 0; e < 16; ++e) {
    float s = fcb[e];
    for (int j = 0; j < 16; ++j) s += n_[j] * fcW[j * 16 + e];
    y[e] = fmaxf(s, 0.f);
  }
  for (int e = 0; e < 16; ++e) t[e] = n_[e] + y[e];
  mu = 0.f; for (int e = 0; e < 16; ++e) mu += t[e]; mu *= (1.f / 16.f);
  var = 0.f; for (int e = 0; e < 16; ++e) { float d = t[e] - mu; var += d * d; }
  inv = rsqrtf(var * (1.f / 16.f) + LN_EPS);
  for (int e = 0; e < 16; ++e) o[e] = (t[e] - mu) * inv * lng[e] + lnb[e];
#pragma unroll
  for (int e = 0; e < 16; ++e) xout[row * 16 + e] = o[e];
}

// ---------------------------------------------------------------------------
extern "C" void kernel_launch(void* const* d_in, const int* in_sizes, int n_in,
                              void* d_out, int out_size, void* d_ws, size_t ws_size,
                              hipStream_t stream)
{
  (void)in_sizes; (void)n_in; (void)out_size; (void)ws_size;
  const float* obj      = (const float*)d_in[0];
  const float* road     = (const float*)d_in[1];
  const float* rg_dW    = (const float*)d_in[2];
  const float* rg_db    = (const float*)d_in[3];
  const float* rg_qkvW  = (const float*)d_in[4];
  const float* rg_qkvB  = (const float*)d_in[5];
  const float* rg_Wq    = (const float*)d_in[6];
  const float* rg_Wk    = (const float*)d_in[7];
  const float* rg_Wv    = (const float*)d_in[8];
  const float* rg_Wo    = (const float*)d_in[9];
  const float* rg_lng   = (const float*)d_in[10];
  const float* rg_lnb   = (const float*)d_in[11];
  const float* rg_fcW   = (const float*)d_in[12];
  const float* rg_fcb   = (const float*)d_in[13];
  const float* rg_outW  = (const float*)d_in[14];
  const float* rg_outb  = (const float*)d_in[15];
  const float* denseW   = (const float*)d_in[16];
  const float* denseb   = (const float*)d_in[17];
  const float* pos      = (const float*)d_in[18];
  const float* Wq_all   = (const float*)d_in[19];
  const float* Wk_all   = (const float*)d_in[20];
  const float* Wv_all   = (const float*)d_in[21];
  const float* Wo_all   = (const float*)d_in[22];
  const float* lng_all  = (const float*)d_in[23];
  const float* lnb_all  = (const float*)d_in[24];
  const float* fcW_all  = (const float*)d_in[25];
  const float* fcb_all  = (const float*)d_in[26];

  uintptr_t w = (uintptr_t)d_ws;
  auto alloc = [&](size_t n) { uintptr_t p = w; w = (w + n + 255) & ~(uintptr_t)255; return p; };

  float* x_rg_a  = (float*)alloc(65536ull * 16 * 4);
  float* x_rg_b  = (float*)alloc(65536ull * 16 * 4);
  float* qp_rg   = (float*)alloc(65536ull * 16 * 4);
  float* v_rg    = (float*)alloc(65536ull * 16 * 4);
  float* rg_part = (float*)alloc(256ull * 80 * 4);
  float* rg_kvk  = (float*)alloc(8ull * 80 * 4);
  u16*   xrg_bf  = (u16*)alloc(65536ull * 32 * 2);
  u16*   rgoWt   = (u16*)alloc(1024ull * 32 * 2);
  u16*   rg_outbf= (u16*)alloc(65536ull * 1024 * 2);
  u16*   obj_bf  = (u16*)alloc(1024ull * 512 * 2);
  u16*   dWt     = (u16*)alloc(1024ull * 512 * 2);
  u16*   Wq_t    = (u16*)alloc(1024ull * 1024 * 2);
  u16*   Wk_t    = (u16*)alloc(1024ull * 1024 * 2);
  u16*   Wv_t    = (u16*)alloc(1024ull * 1024 * 2);
  u16*   Wo_t    = (u16*)alloc(1024ull * 1024 * 2);
  u16*   fcWt    = (u16*)alloc(1024ull * 1024 * 2);
  float* x_f     = (float*)alloc(1024ull * 1024 * 4);
  float* qp_f    = (float*)alloc(1024ull * 1024 * 4);
  float* attn_f  = (float*)alloc(1024ull * 1024 * 4);
  float* n_f     = (float*)alloc(1024ull * 1024 * 4);
  float* y_f     = (float*)alloc(1024ull * 1024 * 4);
  u16*   x_bf    = (u16*)alloc(1024ull * 1024 * 2);
  u16*   n_bf    = (u16*)alloc(1024ull * 1024 * 2);
  u16*   attn_bf = (u16*)alloc(1024ull * 1024 * 2);
  float* kv_part = (float*)alloc(8ull * 16 * 16 * 4096 * 4);
  float* ks_part = (float*)alloc(8ull * 16 * 16 * 64 * 4);
  float* kv_f    = (float*)alloc(8ull * 16 * 4096 * 4);
  float* ks_f    = (float*)alloc(8ull * 16 * 64 * 4);

  auto gemm = [&](const u16* A, const u16* Bt, const float* bias, float* Cf, u16* Cbf,
                  int M, int N, int K, int act, float scale) {
    dim3 g(N / 128, M / 64);
    wmma_gemm_kernel<<<g, 256, 0, stream>>>(A, Bt, bias, Cf, Cbf, M, N, K, act, scale);
  };
  auto tcvt = [&](const float* W, u16* Wt, int K, int N, int Kpad) {
    long long tot = (long long)N * Kpad;
    transpose_cvt_kernel<<<(unsigned)((tot + 255) / 256), 256, 0, stream>>>(W, Wt, K, N, Kpad);
  };

  // ---- one-time weight prep ----
  tcvt(denseW, dWt, 512, 1024, 512);
  tcvt(rg_outW, rgoWt, 16, 1024, 32);
  cvt_pad_kernel<<<(1024 * 512 + 255) / 256, 256, 0, stream>>>(obj, obj_bf, 512, 512, 1024);

  // ---- road-graph encoder ----
  rg_dense_kernel<<<(65536 * 16) / 256, 256, 0, stream>>>(road, rg_dW, rg_db, x_rg_a);
  float* xr_cur = x_rg_a; float* xr_nxt = x_rg_b;
  for (int i = 0; i < 2; ++i) {
    rg_attnA_kernel<<<256, 256, 0, stream>>>(xr_cur, rg_qkvW, rg_qkvB,
        rg_Wq + i * 256, rg_Wk + i * 256, rg_Wv + i * 256, qp_rg, v_rg, rg_part);
    rg_reduce_kernel<<<8, 128, 0, stream>>>(rg_part, rg_kvk);
    rg_attnC_kernel<<<256, 256, 0, stream>>>(xr_cur, qp_rg, rg_kvk, rg_Wo + i * 256,
        rg_lng + i * 16, rg_lnb + i * 16, rg_fcW + i * 256, rg_fcb + i * 16, xr_nxt);
    float* t = xr_cur; xr_cur = xr_nxt; xr_nxt = t;
  }
  // rg_out = relu(x @ rg_out_W + b), bf16, K padded 16->32
  cvt_pad_kernel<<<(65536 * 32) / 256, 256, 0, stream>>>(xr_cur, xrg_bf, 16, 32, 65536);
  gemm(xrg_bf, rgoWt, rg_outb, nullptr, rg_outbf, 65536, 1024, 32, 1, 0.f);

  // ---- object encoder ----
  gemm(obj_bf, dWt, denseb, y_f, nullptr, 1024, 1024, 512, 1, 0.f);
  add_pos_kernel<<<(1024 * 1024) / 256, 256, 0, stream>>>(y_f, pos, x_f, x_bf);

  for (int i = 0; i < 2; ++i) {
    const size_t wofs = (size_t)i * 1024 * 1024;
    tcvt(Wq_all + wofs, Wq_t, 1024, 1024, 1024);
    tcvt(Wk_all + wofs, Wk_t, 1024, 1024, 1024);
    tcvt(Wv_all + wofs, Wv_t, 1024, 1024, 1024);
    tcvt(Wo_all + wofs, Wo_t, 1024, 1024, 1024);
    tcvt(fcW_all + wofs, fcWt, 1024, 1024, 1024);

    // qp = relu(x @ Wq * d^-0.25) + STAB
    gemm(x_bf, Wq_t, nullptr, qp_f, nullptr, 1024, 1024, 1024, 2, OBJ_NORM);
    // fused K/V projection + kv/ksum reduction over S=8192
    {
      dim3 g(16, 16, 8);
      kv_partial_kernel<<<g, 256, 0, stream>>>(rg_outbf, Wk_t, Wv_t, kv_part, ks_part, OBJ_NORM);
    }
    kv_reduce_kernel<<<128, 256, 0, stream>>>(kv_part, ks_part, kv_f, ks_f);
    attn_combine_kernel<<<1024, 256, 0, stream>>>(qp_f, kv_f, ks_f, attn_bf);
    gemm(attn_bf, Wo_t, nullptr, attn_f, nullptr, 1024, 1024, 1024, 0, 0.f);
    ln_res_kernel<<<1024, 256, 0, stream>>>(attn_f, x_f, lng_all + i * 1024,
                                            lnb_all + i * 1024, n_f, n_bf);
    gemm(n_bf, fcWt, fcb_all + i * 1024, y_f, nullptr, 1024, 1024, 1024, 1, 0.f);
    float* xo = (i == 1) ? (float*)d_out : x_f;
    ln_res_kernel<<<1024, 256, 0, stream>>>(n_f, y_f, lng_all + i * 1024,
                                            lnb_all + i * 1024, xo, x_bf);
  }
}
